// Block_51505247813860
// MI455X (gfx1250) — compile-verified, matrix operand order checked by reading, not run on hardware
//
#include <hip/hip_runtime.h>
#include <hip/hip_bf16.h>

typedef unsigned short ushort_t;
typedef __attribute__((ext_vector_type(16))) __bf16 v16bf;
typedef __attribute__((ext_vector_type(8)))  float  v8f;

__device__ __forceinline__ ushort_t f2bf(float f) {
    unsigned int u = __float_as_uint(f);
    unsigned int r = u + 0x7FFFu + ((u >> 16) & 1u);   // round-to-nearest-even
    return (ushort_t)(r >> 16);
}

// ---- CDNA5 async global->LDS copy (ASYNCcnt-tracked, no VGPR staging) ----
// 16 bytes per lane. LDS address = low 32 bits of the generic shared pointer
// (flat-LDS addressing truncates to addr[31:0], same convention as ds_* ops).
__device__ __forceinline__ void async_copy16(const ushort_t* g, ushort_t* l) {
    asm volatile("global_load_async_to_lds_b128 %0, %1, off"
                 :: "v"((unsigned)(size_t)l), "v"((unsigned long long)(size_t)g)
                 : "memory");
}
__device__ __forceinline__ void wait_async0() {
    asm volatile("s_wait_asynccnt 0x0" ::: "memory");
}

// Build a 16x32 (A) or 32x16 (B) bf16 WMMA fragment from row-major storage.
// Per CDNA5 ISA layout: lanes 0-15 hold K = {0..7, 16..23}, lanes 16-31 hold
// K = {8..15, 24..31}; i.e. two contiguous 8-element chunks per lane.
// p points at &buf[row*ld + kbase]; lh = lane>>4.
__device__ __forceinline__ v16bf load_frag(const ushort_t* p, int lh) {
    union { uint4 q[2]; v16bf f; } u;
    u.q[0] = *(const uint4*)(p + lh * 8);
    u.q[1] = *(const uint4*)(p + 16 + lh * 8);
    return u.f;
}

// ---------------------------------------------------------------- utilities
__global__ void cvt_bf16_kernel(const float* __restrict__ in,
                                ushort_t* __restrict__ out, size_t n) {
    size_t i = (size_t)blockIdx.x * blockDim.x + threadIdx.x;
    if (i < n) out[i] = f2bf(in[i]);
}

__global__ void tail_zero_kernel(float* p) { *p = 0.0f; }

// ---------------------------------------------------------------- layernorm
__global__ __launch_bounds__(256)
void ln_kernel(const float* __restrict__ x, const float* __restrict__ w,
               const float* __restrict__ b, ushort_t* __restrict__ out, int C) {
    __shared__ float red[256];
    const int row = blockIdx.x;
    const float* xr = x + (size_t)row * C;

    float s = 0.f;
    for (int i = threadIdx.x; i < C; i += 256) s += xr[i];
    red[threadIdx.x] = s; __syncthreads();
    for (int st = 128; st > 0; st >>= 1) {
        if (threadIdx.x < st) red[threadIdx.x] += red[threadIdx.x + st];
        __syncthreads();
    }
    const float mu = red[0] / (float)C;
    __syncthreads();

    float v = 0.f;
    for (int i = threadIdx.x; i < C; i += 256) { float d = xr[i] - mu; v += d * d; }
    red[threadIdx.x] = v; __syncthreads();
    for (int st = 128; st > 0; st >>= 1) {
        if (threadIdx.x < st) red[threadIdx.x] += red[threadIdx.x + st];
        __syncthreads();
    }
    const float rstd = rsqrtf(red[0] / (float)C + 1e-5f);

    for (int i = threadIdx.x; i < C; i += 256)
        out[(size_t)row * C + i] = f2bf((xr[i] - mu) * rstd * w[i] + b[i]);
}

// ---------------------------------------------------------------- GEMM
// out[m,n] = epilogue( sum_k A[m,k] * W[n,k] )  -- A:[M,K] bf16, W:[N,K] bf16
// 128x128x32 tiles, double-buffered LDS filled by async global->LDS copies.
enum { EPI_BF16 = 0, EPI_GELU_BF16 = 1, EPI_RESID_F32 = 2 };

template <int EPI>
__global__ __launch_bounds__(256)
void gemm_bf16_kernel(const ushort_t* __restrict__ A, const ushort_t* __restrict__ W,
                      int M, int N, int K,
                      const float* __restrict__ bias,
                      const float* __restrict__ resid,
                      void* __restrict__ outp) {
    constexpr int LDSS = 48;                     // 32 cols + pad, 96B rows (16B aligned)
    __shared__ __align__(16) ushort_t As[2][128 * LDSS];
    __shared__ __align__(16) ushort_t Ws[2][128 * LDSS];

    const int tid  = threadIdx.x;
    const int wave = tid >> 5;                   // wave32
    const int lane = tid & 31;
    const int lh   = lane >> 4;
    const int ll   = lane & 15;
    const int m0   = blockIdx.y * 128;
    const int n0   = blockIdx.x * 128;
    const int wm   = wave & 1;                   // 2 row-strips of 64
    const int wn   = wave >> 1;                  // 4 col-strips of 32

    v8f zero = {};
    v8f acc[4][2];
#pragma unroll
    for (int i = 0; i < 4; ++i)
#pragma unroll
        for (int j = 0; j < 2; ++j) acc[i][j] = zero;

    const int ldr  = tid >> 1;                   // 0..127 staging row
    const int lcol = (tid & 1) << 4;             // 0 or 16

    // async stage of one 128x32 A tile and one 128x32 W tile (32B/thread each)
    auto stage = [&](int kt, int buf) {
        const ushort_t* ga = A + (size_t)(m0 + ldr) * K + kt * 32 + lcol;
        const ushort_t* gw = W + (size_t)(n0 + ldr) * K + kt * 32 + lcol;
        ushort_t* la = &As[buf][ldr * LDSS + lcol];
        ushort_t* lw = &Ws[buf][ldr * LDSS + lcol];
        async_copy16(ga,     la);
        async_copy16(ga + 8, la + 8);
        async_copy16(gw,     lw);
        async_copy16(gw + 8, lw + 8);
    };

    const int nk = K >> 5;
    stage(0, 0);
    wait_async0();
    __syncthreads();

    for (int kt = 0; kt < nk; ++kt) {
        const int cur = kt & 1;
        if (kt + 1 < nk) stage(kt + 1, cur ^ 1);   // overlap with compute below

        v16bf bfr[2];
#pragma unroll
        for (int j = 0; j < 2; ++j)
            bfr[j] = load_frag(&Ws[cur][(wn * 32 + j * 16 + ll) * LDSS], lh);
#pragma unroll
        for (int i = 0; i < 4; ++i) {
            v16bf afr = load_frag(&As[cur][(wm * 64 + i * 16 + ll) * LDSS], lh);
#pragma unroll
            for (int j = 0; j < 2; ++j)
                acc[i][j] = __builtin_amdgcn_wmma_f32_16x16x32_bf16(
                    false, afr, false, bfr[j], (short)0, acc[i][j], false, false);
        }

        wait_async0();                             // next tile landed in LDS
        __syncthreads();
    }

    // epilogue: element (m, n) = (m0 + wm*64 + i*16 + r + 8*lh, n0 + wn*32 + j*16 + ll)
#pragma unroll
    for (int i = 0; i < 4; ++i)
#pragma unroll
        for (int j = 0; j < 2; ++j) {
            const int col = n0 + wn * 32 + j * 16 + ll;
#pragma unroll
            for (int r = 0; r < 8; ++r) {
                const int row = m0 + wm * 64 + i * 16 + r + 8 * lh;
                float v = acc[i][j][r];
                if (EPI == EPI_GELU_BF16) {
                    v += bias[col];
                    v = 0.5f * v * (1.0f + erff(v * 0.7071067811865475f));
                    ((ushort_t*)outp)[(size_t)row * N + col] = f2bf(v);
                } else if (EPI == EPI_RESID_F32) {
                    if (bias) v += bias[col];
                    v += resid[(size_t)row * N + col];
                    ((float*)outp)[(size_t)row * N + col] = v;
                } else {
                    ((ushort_t*)outp)[(size_t)row * N + col] = f2bf(v);
                }
            }
        }
}

// ---------------------------------------------------------------- flash attention
// qkv: [B*T, 3C] bf16 (q | k | v per row).  y: [B*T, C] bf16.
// grid: (T/128, B*NH); block 256 = 8 waves; wave w handles 16 query rows.
__global__ __launch_bounds__(256)
void flash_kernel(const ushort_t* __restrict__ qkv, ushort_t* __restrict__ y) {
    constexpr int T = 2048, C = 1024, HD = 64;
    constexpr int LDQ = 3 * C;
    __shared__ __align__(16) ushort_t kt[32 * 64];        // K tile, row-major [key][d]
    __shared__ __align__(16) ushort_t vt[64 * 32];        // V tile transposed [d][key]
    __shared__ __align__(16) ushort_t pt[8 * 16 * 32];    // per-wave P scratch

    const int tid  = threadIdx.x;
    const int wave = tid >> 5;
    const int lane = tid & 31;
    const int lh   = lane >> 4;
    const int ll   = lane & 15;
    const int b    = blockIdx.y >> 4;
    const int head = blockIdx.y & 15;
    const int qrow0 = blockIdx.x * 128 + wave * 16;
    const size_t rowbase = (size_t)b * T;

    // Q fragments (16 rows x 64 d), straight from global
    v16bf qf[2];
#pragma unroll
    for (int f = 0; f < 2; ++f)
        qf[f] = load_frag(qkv + (rowbase + qrow0 + ll) * LDQ + head * HD + f * 32, lh);

    v8f zero = {};
    v8f o[4];
#pragma unroll
    for (int t = 0; t < 4; ++t) o[t] = zero;
    float mrow[8], lrow[8];
#pragma unroll
    for (int r = 0; r < 8; ++r) { mrow[r] = -1e30f; lrow[r] = 0.f; }

    ushort_t* ptw = pt + wave * (16 * 32);
    const int qhi = blockIdx.x * 128 + 128;

    for (int j0 = 0; j0 < qhi; j0 += 32) {
        {   // stage K asynchronously (row-major); V via VGPR + transpose scatter
            const int krow = tid >> 3;
            const int c8   = (tid & 7) << 3;
            const size_t src = (rowbase + j0 + krow) * LDQ + head * HD + c8;
            async_copy16(qkv + src + C, &kt[krow * 64 + c8]);
            union { uint4 v; ushort_t u[8]; } vv;
            vv.v = *(const uint4*)(qkv + src + 2 * C);
#pragma unroll
            for (int i = 0; i < 8; ++i) vt[(c8 + i) * 32 + krow] = vv.u[i];
        }
        wait_async0();
        __syncthreads();

        // S = Q @ K^T (16x32), two 16x16 tiles, K-dim = 64 -> 2 wmma each
        v8f s[2]; s[0] = zero; s[1] = zero;
#pragma unroll
        for (int cg = 0; cg < 2; ++cg)
#pragma unroll
            for (int f = 0; f < 2; ++f) {
                v16bf kb = load_frag(&kt[(cg * 16 + ll) * 64 + f * 32], lh);
                s[cg] = __builtin_amdgcn_wmma_f32_16x16x32_bf16(
                    false, qf[f], false, kb, (short)0, s[cg], false, false);
            }

        // scale + causal mask
#pragma unroll
        for (int cg = 0; cg < 2; ++cg)
#pragma unroll
            for (int r = 0; r < 8; ++r) {
                const int col = j0 + cg * 16 + ll;
                const int row = qrow0 + r + 8 * lh;
                const float v = s[cg][r] * 0.125f;       // hd^-0.5, hd=64
                s[cg][r] = (col <= row) ? v : -1e30f;
            }

        // online softmax (row lives across the 16 lanes of each half)
#pragma unroll
        for (int r = 0; r < 8; ++r) {
            float mx = fmaxf(s[0][r], s[1][r]);
#pragma unroll
            for (int d = 8; d >= 1; d >>= 1) mx = fmaxf(mx, __shfl_xor(mx, d, 32));
            const float mnew  = fmaxf(mrow[r], mx);
            const float alpha = __expf(mrow[r] - mnew);
            const float p0 = __expf(s[0][r] - mnew);
            const float p1 = __expf(s[1][r] - mnew);
            s[0][r] = p0; s[1][r] = p1;
            float rs = p0 + p1;
#pragma unroll
            for (int d = 8; d >= 1; d >>= 1) rs += __shfl_xor(rs, d, 32);
            lrow[r] = lrow[r] * alpha + rs;
            mrow[r] = mnew;
#pragma unroll
            for (int t = 0; t < 4; ++t) o[t][r] *= alpha;
        }

        // P: C-layout -> A-fragment layout via per-wave LDS round trip
#pragma unroll
        for (int cg = 0; cg < 2; ++cg)
#pragma unroll
            for (int r = 0; r < 8; ++r)
                ptw[(r + 8 * lh) * 32 + cg * 16 + ll] = f2bf(s[cg][r]);
        v16bf pa = load_frag(&ptw[ll * 32], lh);

        // O += P @ V   (V^T staged: column d of V is a contiguous vt row)
#pragma unroll
        for (int t = 0; t < 4; ++t) {
            v16bf vb = load_frag(&vt[(t * 16 + ll) * 32], lh);
            o[t] = __builtin_amdgcn_wmma_f32_16x16x32_bf16(
                false, pa, false, vb, (short)0, o[t], false, false);
        }
        __syncthreads();
    }

    // normalize and store
#pragma unroll
    for (int t = 0; t < 4; ++t)
#pragma unroll
        for (int r = 0; r < 8; ++r) {
            const int row = qrow0 + r + 8 * lh;
            const int col = head * HD + t * 16 + ll;
            y[(rowbase + row) * C + col] = f2bf(o[t][r] * (1.0f / lrow[r]));
        }
}

// ---------------------------------------------------------------- launcher
extern "C" void kernel_launch(void* const* d_in, const int* in_sizes, int n_in,
                              void* d_out, int out_size, void* d_ws, size_t ws_size,
                              hipStream_t stream) {
    (void)in_sizes; (void)n_in; (void)out_size; (void)ws_size;
    const float* x      = (const float*)d_in[0];
    const float* ln1_w  = (const float*)d_in[1];
    const float* ln1_b  = (const float*)d_in[2];
    const float* ln2_w  = (const float*)d_in[3];
    const float* ln2_b  = (const float*)d_in[4];
    const float* qkv_w  = (const float*)d_in[5];
    const float* proj_w = (const float*)d_in[6];
    const float* ff_w1  = (const float*)d_in[7];
    const float* ff_b1  = (const float*)d_in[8];
    const float* ff_w2  = (const float*)d_in[9];
    const float* ff_b2  = (const float*)d_in[10];

    constexpr int Bb = 4, T = 2048, C = 1024, NH = 16, DFF = 4096;
    constexpr int Mr = Bb * T;                   // 8192 rows

    char* ws = (char*)d_ws;
    size_t off = 0;
    auto take = [&](size_t bytes) -> char* {
        char* p = ws + off;
        off += (bytes + 255) & ~(size_t)255;
        return p;
    };
    ushort_t* wq   = (ushort_t*)take((size_t)3 * C * C * 2);
    ushort_t* wp   = (ushort_t*)take((size_t)C * C * 2);
    ushort_t* w1   = (ushort_t*)take((size_t)DFF * C * 2);
    ushort_t* w2   = (ushort_t*)take((size_t)C * DFF * 2);
    ushort_t* hbuf = (ushort_t*)take((size_t)Mr * C * 2);        // h, then y_att
    ushort_t* qkvb = (ushort_t*)take((size_t)Mr * 3 * C * 2);    // qkv, then h2
    float*    x1   = (float*)   take((size_t)Mr * C * 4);
    ushort_t* fbuf = (ushort_t*)take((size_t)Mr * DFF * 2);

    auto cvt = [&](const float* in, ushort_t* out, size_t n) {
        cvt_bf16_kernel<<<(unsigned)((n + 255) / 256), 256, 0, stream>>>(in, out, n);
    };
    cvt(qkv_w,  wq, (size_t)3 * C * C);
    cvt(proj_w, wp, (size_t)C * C);
    cvt(ff_w1,  w1, (size_t)DFF * C);
    cvt(ff_w2,  w2, (size_t)C * DFF);

    // 1) h = LN1(x)
    ln_kernel<<<Mr, 256, 0, stream>>>(x, ln1_w, ln1_b, hbuf, C);
    // 2) qkv = h @ qkv_w^T
    gemm_bf16_kernel<EPI_BF16><<<dim3(3 * C / 128, Mr / 128), 256, 0, stream>>>(
        hbuf, wq, Mr, 3 * C, C, nullptr, nullptr, qkvb);
    // 3) y = causal attention(q, k, v)   (writes into hbuf)
    flash_kernel<<<dim3(T / 128, Bb * NH), 256, 0, stream>>>(qkvb, hbuf);
    // 4) x1 = x + y @ proj_w^T
    gemm_bf16_kernel<EPI_RESID_F32><<<dim3(C / 128, Mr / 128), 256, 0, stream>>>(
        hbuf, wp, Mr, C, C, nullptr, x, x1);
    // 5) h2 = LN2(x1)   (reuse qkv buffer)
    ln_kernel<<<Mr, 256, 0, stream>>>(x1, ln2_w, ln2_b, qkvb, C);
    // 6) f = gelu(h2 @ ff_w1^T + b1)
    gemm_bf16_kernel<EPI_GELU_BF16><<<dim3(DFF / 128, Mr / 128), 256, 0, stream>>>(
        qkvb, w1, Mr, DFF, C, ff_b1, nullptr, fbuf);
    // 7) out = x1 + f @ ff_w2^T + b2
    gemm_bf16_kernel<EPI_RESID_F32><<<dim3(C / 128, Mr / 128), 256, 0, stream>>>(
        fbuf, w2, Mr, C, DFF, ff_b2, x1, (float*)d_out);
    // 8) trailing scalar output = 0
    tail_zero_kernel<<<1, 1, 0, stream>>>((float*)d_out + (size_t)Mr * C);
}
